// BaseAttention_77962246357243
// MI455X (gfx1250) — matrix operand order repaired and checked by reading, not compiled, not run on hardware
//
#include <hip/hip_runtime.h>
#include <hip/hip_bf16.h>

// ---------------------------------------------------------------------------
// Problem constants: B=4, T=2048, D=1024, H=16, HD=64
// ---------------------------------------------------------------------------
#define BATCH 4
#define SEQ   2048
#define DIM   1024
#define NH    16
#define HD    64
#define M_ROWS (BATCH * SEQ)    // 8192
#define SCALE 0.125f            // 64^-0.5

typedef __attribute__((ext_vector_type(16))) __bf16        v16bf;
typedef __attribute__((ext_vector_type(8)))  float         v8f;
typedef __attribute__((ext_vector_type(8)))  unsigned int  v8u;

union frag16 { v8u u; v16bf b; };

__device__ __forceinline__ unsigned short f2bf(float f) {
  unsigned int u = __float_as_uint(f);
  unsigned int r = u + 0x7FFFu + ((u >> 16) & 1u);   // RNE
  return (unsigned short)(r >> 16);
}

// A-fragment, 16-bit 16x32 (M x K), row-major source, ld in elements.
// ISA 7.12.2: lane m=lane&15; kh=lane>>4; VGPR i holds K=(i>>2)*16 + kh*8 + (i&3)*2
__device__ __forceinline__ v16bf load_frag_a(const unsigned short* p0, int ld) {
  int lane = threadIdx.x & 31;
  int r = lane & 15, kh = lane >> 4;
  const unsigned short* p = p0 + (size_t)r * ld + kh * 8;
  frag16 f;
#pragma unroll
  for (int i = 0; i < 8; ++i) {
    int k = ((i >> 2) << 4) + ((i & 3) << 1);
    f.u[i] = *(const unsigned int*)(p + k);
  }
  return f.b;
}

// B-fragment, 16-bit 32x16 (K x N), source stored as [N][K] row-major (ld elems).
// Same K order as A: VGPR i holds K=(i>>2)*16 + kh*8 + (i&3)*2 (+1).
__device__ __forceinline__ v16bf load_frag_b(const unsigned short* p0, int ld) {
  int lane = threadIdx.x & 31;
  int c = lane & 15, kh = lane >> 4;
  const unsigned short* p = p0 + (size_t)c * ld + kh * 8;
  frag16 f;
#pragma unroll
  for (int i = 0; i < 8; ++i) {
    f.u[i] = *(const unsigned int*)(p + ((i >> 2) << 4) + ((i & 3) << 1));
  }
  return f.b;
}

__device__ __forceinline__ v8f wmma_bf16(v16bf a, v16bf b, v8f c) {
  return __builtin_amdgcn_wmma_f32_16x16x32_bf16(false, a, false, b, (short)0, c,
                                                 false, false);
}

// ---------------------------------------------------------------------------
// Kernel 0: f32 -> bf16 conversion
// ---------------------------------------------------------------------------
__global__ __launch_bounds__(256) void cvt_bf16_kernel(const float* __restrict__ src,
                                                       unsigned short* __restrict__ dst,
                                                       int n) {
  int i = blockIdx.x * 256 + threadIdx.x;
  if (i < n) dst[i] = f2bf(src[i]);
}

// ---------------------------------------------------------------------------
// Kernel 1: QKV projection  y[m,n] = sum_k x[m,k] * w[n,k]   (x @ W^T)
// Register tiling: each wave computes 32(M) x 64(N) = 2x4 WMMA tiles.
// Block = 8 waves arranged 4(M) x 2(N) -> 128 x 128 block tile.
// Scatter into q[BH][T][HD], k[BH][T][HD], vT[BH][HD][T]  (all bf16)
// ---------------------------------------------------------------------------
__global__ __launch_bounds__(256) void qkv_gemm_kernel(const unsigned short* __restrict__ xb,
                                                       const unsigned short* __restrict__ wb,
                                                       unsigned short* __restrict__ qb,
                                                       unsigned short* __restrict__ kb,
                                                       unsigned short* __restrict__ vtb) {
  int wave  = threadIdx.x >> 5;
  int bm    = blockIdx.x & 63;           // 64 M-blocks of 128 rows
  int bn    = blockIdx.x >> 6;           // 24 N-blocks of 128 cols
  int mbase = bm * 128 + (wave >> 1) * 32;
  int nbase = bn * 128 + (wave & 1) * 64;

  const unsigned short* ap0 = xb + (size_t)mbase * DIM;
  const unsigned short* ap1 = ap0 + (size_t)16 * DIM;
  const unsigned short* bp  = wb + (size_t)nbase * DIM;

  v8f acc[2][4];
#pragma unroll
  for (int i = 0; i < 2; ++i)
#pragma unroll
    for (int j = 0; j < 4; ++j) acc[i][j] = (v8f){};

  for (int kk = 0; kk < DIM; kk += 32) {
    __builtin_prefetch(ap0 + kk + 256, 0, 3);   // global_prefetch_b8 (near)
    __builtin_prefetch(bp  + kk + 256, 0, 3);
    v16bf a0 = load_frag_a(ap0 + kk, DIM);
    v16bf a1 = load_frag_a(ap1 + kk, DIM);
#pragma unroll
    for (int j = 0; j < 4; ++j) {
      v16bf b = load_frag_b(bp + (size_t)j * 16 * DIM + kk, DIM);
      acc[0][j] = wmma_bf16(a0, b, acc[0][j]);
      acc[1][j] = wmma_bf16(a1, b, acc[1][j]);
    }
  }

  int lane = threadIdx.x & 31;
  int col  = lane & 15, kh = lane >> 4;
#pragma unroll
  for (int j = 0; j < 4; ++j) {
    int n = nbase + j * 16 + col;
    int which = n >> 10;                 // 0=q 1=k 2=v (16-col tile never crosses)
    int r10   = n & 1023;
    int h = r10 >> 6, d = r10 & 63;
#pragma unroll
    for (int qi = 0; qi < 2; ++qi) {
#pragma unroll
      for (int i = 0; i < 8; ++i) {
        int m = mbase + qi * 16 + i + 8 * kh;
        int b_ = m >> 11;                // / SEQ
        int t  = m & (SEQ - 1);
        size_t bh = (size_t)b_ * NH + h;
        unsigned short val = f2bf(acc[qi][j][i]);
        if (which == 0)      qb[(bh * SEQ + t) * HD + d]  = val;
        else if (which == 1) kb[(bh * SEQ + t) * HD + d]  = val;
        else                 vtb[(bh * HD + d) * SEQ + t] = val;  // V transposed
      }
    }
  }
}

// ---------------------------------------------------------------------------
// Kernel 2: causal flash attention. One wave per (bh, 32-query-row tile);
// every K/V fragment is reused by both 16-row halves (2x fragment reuse).
// Keys streamed 32 at a time: 8 WMMA for Q.K^T, online softmax, 8 WMMA for P.V.
// ---------------------------------------------------------------------------
__global__ __launch_bounds__(256) void attention_kernel(const unsigned short* __restrict__ qb,
                                                        const unsigned short* __restrict__ kb,
                                                        const unsigned short* __restrict__ vtb,
                                                        unsigned short* __restrict__ attnb) {
  __shared__ __align__(16) unsigned short pbuf[8][2][16 * 32];  // per-wave P tiles

  int wave = threadIdx.x >> 5;
  int lane = threadIdx.x & 31;
  int gw   = blockIdx.x * 8 + wave;      // 0..4095
  int bh   = gw >> 6;                    // 64 (b*16 + h)
  int qt   = gw & 63;                    // 64 tiles of 32 query rows
  int qbase = qt * 32;

  const unsigned short* qp = qb  + ((size_t)bh * SEQ + qbase) * HD;
  const unsigned short* kp = kb  + (size_t)bh * SEQ * HD;
  const unsigned short* vp = vtb + (size_t)bh * HD * SEQ;

  v16bf aq[2][2];
#pragma unroll
  for (int qi = 0; qi < 2; ++qi) {
    aq[qi][0] = load_frag_a(qp + (size_t)qi * 16 * HD, HD);
    aq[qi][1] = load_frag_a(qp + (size_t)qi * 16 * HD + 32, HD);
  }

  v8f o[2][4];
  float mrow[2][8], lrow[2][8];
#pragma unroll
  for (int qi = 0; qi < 2; ++qi) {
#pragma unroll
    for (int t = 0; t < 4; ++t) o[qi][t] = (v8f){};
#pragma unroll
    for (int i = 0; i < 8; ++i) { mrow[qi][i] = -1e30f; lrow[qi][i] = 0.0f; }
  }

  int col = lane & 15, kh = lane >> 4;
  int jmax = qbase + 32;                  // causal: keys < qbase+32

  for (int jb = 0; jb < jmax; jb += 32) {
    // ---- scores: 32 queries x 32 keys --------------------------------------
    v16bf bk0 = load_frag_b(kp + (size_t)jb * HD, HD);
    v16bf bk1 = load_frag_b(kp + (size_t)jb * HD + 32, HD);
    v16bf bk2 = load_frag_b(kp + (size_t)(jb + 16) * HD, HD);
    v16bf bk3 = load_frag_b(kp + (size_t)(jb + 16) * HD + 32, HD);
    v8f z = {};
    v8f s[2][2];
#pragma unroll
    for (int qi = 0; qi < 2; ++qi) {
      s[qi][0] = wmma_bf16(aq[qi][1], bk1, wmma_bf16(aq[qi][0], bk0, z));
      s[qi][1] = wmma_bf16(aq[qi][1], bk3, wmma_bf16(aq[qi][0], bk2, z));
    }

    int j0 = jb + col, j1 = j0 + 16;
#pragma unroll
    for (int qi = 0; qi < 2; ++qi) {
#pragma unroll
      for (int i = 0; i < 8; ++i) {
        int qrow = qbase + qi * 16 + i + 8 * kh;
        float x0 = s[qi][0][i] * SCALE; if (j0 > qrow) x0 = -1e30f;
        float x1 = s[qi][1][i] * SCALE; if (j1 > qrow) x1 = -1e30f;

        float mx = fmaxf(x0, x1);
#pragma unroll
        for (int off = 1; off < 16; off <<= 1)
          mx = fmaxf(mx, __shfl_xor(mx, off, 32));   // reduce within 16-lane group

        float mnew  = fmaxf(mrow[qi][i], mx);
        float alpha = __expf(mrow[qi][i] - mnew);
        float p0 = __expf(x0 - mnew);
        float p1 = __expf(x1 - mnew);
        float rs = p0 + p1;
#pragma unroll
        for (int off = 1; off < 16; off <<= 1)
          rs += __shfl_xor(rs, off, 32);

        lrow[qi][i] = lrow[qi][i] * alpha + rs;
        mrow[qi][i] = mnew;
#pragma unroll
        for (int t = 0; t < 4; ++t) o[qi][t][i] = o[qi][t][i] * alpha;

        int prow = i + 8 * kh;                       // C layout -> LDS row-major
        pbuf[wave][qi][prow * 32 + col]      = f2bf(p0);
        pbuf[wave][qi][prow * 32 + col + 16] = f2bf(p1);
      }
    }

    // LDS ops of one wave are in-order; stop compiler reordering + drain DScnt
    asm volatile("s_wait_dscnt 0" ::: "memory");

    v16bf ap0 = load_frag_a(&pbuf[wave][0][0], 32);  // P tiles as A-fragments
    v16bf ap1 = load_frag_a(&pbuf[wave][1][0], 32);

#pragma unroll
    for (int t = 0; t < 4; ++t) {
      v16bf bv = load_frag_b(vp + (size_t)(t * 16) * SEQ + jb, SEQ);
      o[0][t] = wmma_bf16(ap0, bv, o[0][t]);
      o[1][t] = wmma_bf16(ap1, bv, o[1][t]);
    }
  }

  // ---- epilogue: normalize and write [B][T][H*HD] ---------------------------
  int b_ = bh >> 4;
  int h  = bh & 15;
#pragma unroll
  for (int qi = 0; qi < 2; ++qi) {
#pragma unroll
    for (int i = 0; i < 8; ++i) {
      float inv = 1.0f / lrow[qi][i];
      int trow = qbase + qi * 16 + i + 8 * kh;
      size_t rowoff = ((size_t)b_ * SEQ + trow) * DIM + h * HD;
#pragma unroll
      for (int t = 0; t < 4; ++t)
        attnb[rowoff + t * 16 + col] = f2bf(o[qi][t][i] * inv);
    }
  }
}

// ---------------------------------------------------------------------------
// Kernel 3: output projection  out[m,n] = sum_k attn[m,k]*out_w[n,k] + bias[n]
// Same 2x4 register tiling as the QKV GEMM; f32 result straight to d_out.
// ---------------------------------------------------------------------------
__global__ __launch_bounds__(256) void out_proj_kernel(const unsigned short* __restrict__ ab,
                                                       const unsigned short* __restrict__ wb,
                                                       const float* __restrict__ bias,
                                                       float* __restrict__ out) {
  int wave  = threadIdx.x >> 5;
  int bm    = blockIdx.x & 63;           // 64 M-blocks of 128
  int bn    = blockIdx.x >> 6;           // 8  N-blocks of 128
  int mbase = bm * 128 + (wave >> 1) * 32;
  int nbase = bn * 128 + (wave & 1) * 64;

  const unsigned short* ap0 = ab + (size_t)mbase * DIM;
  const unsigned short* ap1 = ap0 + (size_t)16 * DIM;
  const unsigned short* bp  = wb + (size_t)nbase * DIM;

  v8f acc[2][4];
#pragma unroll
  for (int i = 0; i < 2; ++i)
#pragma unroll
    for (int j = 0; j < 4; ++j) acc[i][j] = (v8f){};

  for (int kk = 0; kk < DIM; kk += 32) {
    __builtin_prefetch(ap0 + kk + 256, 0, 3);
    __builtin_prefetch(bp  + kk + 256, 0, 3);
    v16bf a0 = load_frag_a(ap0 + kk, DIM);
    v16bf a1 = load_frag_a(ap1 + kk, DIM);
#pragma unroll
    for (int j = 0; j < 4; ++j) {
      v16bf b = load_frag_b(bp + (size_t)j * 16 * DIM + kk, DIM);
      acc[0][j] = wmma_bf16(a0, b, acc[0][j]);
      acc[1][j] = wmma_bf16(a1, b, acc[1][j]);
    }
  }

  int lane = threadIdx.x & 31;
  int col  = lane & 15, kh = lane >> 4;
#pragma unroll
  for (int j = 0; j < 4; ++j) {
    int n = nbase + j * 16 + col;
    float bv = bias[n];
#pragma unroll
    for (int qi = 0; qi < 2; ++qi) {
#pragma unroll
      for (int i = 0; i < 8; ++i) {
        int m = mbase + qi * 16 + i + 8 * kh;
        out[(size_t)m * DIM + n] = acc[qi][j][i] + bv;
      }
    }
  }
}

// ---------------------------------------------------------------------------
// Launch
// ---------------------------------------------------------------------------
extern "C" void kernel_launch(void* const* d_in, const int* in_sizes, int n_in,
                              void* d_out, int out_size, void* d_ws, size_t ws_size,
                              hipStream_t stream) {
  const float* x     = (const float*)d_in[0];   // [4,2048,1024]
  const float* qkv_w = (const float*)d_in[1];   // [3072,1024]
  const float* out_w = (const float*)d_in[2];   // [1024,1024]
  const float* out_b = (const float*)d_in[3];   // [1024]
  float* out = (float*)d_out;

  char* ws = (char*)d_ws;
  // bf16 workspace layout (bytes)
  unsigned short* xb    = (unsigned short*)(ws);                       // 16 MB
  unsigned short* wqkvb = (unsigned short*)(ws + 16777216);            //  6 MB
  unsigned short* wob   = (unsigned short*)(ws + 23068672);            //  2 MB
  unsigned short* qb    = (unsigned short*)(ws + 25165824);            // 16 MB
  unsigned short* kb    = (unsigned short*)(ws + 41943040);            // 16 MB
  unsigned short* vtb   = (unsigned short*)(ws + 58720256);            // 16 MB
  unsigned short* attnb = (unsigned short*)(ws + 75497472);            // 16 MB

  int nx = M_ROWS * DIM;           // 8388608
  int nw = 3 * DIM * DIM;          // 3145728
  int no = DIM * DIM;              // 1048576
  cvt_bf16_kernel<<<(nx + 255) / 256, 256, 0, stream>>>(x, xb, nx);
  cvt_bf16_kernel<<<(nw + 255) / 256, 256, 0, stream>>>(qkv_w, wqkvb, nw);
  cvt_bf16_kernel<<<(no + 255) / 256, 256, 0, stream>>>(out_w, wob, no);

  // (8192/128) * (3072/128) = 64*24 blocks
  qkv_gemm_kernel<<<1536, 256, 0, stream>>>(xb, wqkvb, qb, kb, vtb);

  // 64 bh * 64 q-tiles(32 rows) = 4096 waves -> 512 blocks of 8 waves
  attention_kernel<<<512, 256, 0, stream>>>(qb, kb, vtb, attnb);

  // (8192/128) * (1024/128) = 64*8 blocks
  out_proj_kernel<<<512, 256, 0, stream>>>(attnb, wob, out_b, out);
}
